// StructuralSBNN_53188874993817
// MI455X (gfx1250) — matrix operand order repaired and checked by reading, not compiled, not run on hardware
//
#include <hip/hip_runtime.h>

// ---------------- problem constants ----------------
constexpr int kB    = 32768;
constexpr int kIN   = 512;
constexpr int kHID  = 2048;
constexpr int kLAT  = 128;
constexpr int kEH   = 512;
constexpr int kMID  = 1024;
constexpr int kHALF = 256;
constexpr int kZK   = kLAT + kIN;   // 640
constexpr float kEPS  = 1e-5f;
constexpr float kLN10 = 2.302585092994046f;

// ---------------- vector types ----------------
typedef __attribute__((ext_vector_type(16))) __bf16 bf16x16;
typedef __attribute__((ext_vector_type(8)))  __bf16 bf16x8;
typedef __attribute__((ext_vector_type(4)))  __bf16 bf16x4;
typedef __attribute__((ext_vector_type(8)))  float  f32x8;

__device__ __forceinline__ float eluf(float v) {
    return v > 0.f ? v : expm1f(v);
}

// ============================================================
// WMMA GEMM:  C[M,N] = A[M,K] (bf16, row-major, lda)
//                    x W[N,K]^T (bf16, row-major, ldw)  + bias[N]
// Block: 256 threads (8 waves), tile 128(M) x 128(N).
// Waves: 4 in M x 2 in N; each wave 32x64 = 2x4 WMMA 16x16 tiles.
// MODE 0: write f32 (bias only)          -> BN layers
// MODE 1: write bf16 with bias + ELU     -> expert-head layers
// MODE 2: write bf16 with bias only      -> latent layer
// Requires: M%128==0, N%128==0, K%32==0.
// ============================================================
template <int MODE>
__global__ void __launch_bounds__(256)
gemm_bf16_wmma(const __bf16* __restrict__ A, int lda,
               const __bf16* __restrict__ W, int ldw,
               const float* __restrict__ bias,
               void* __restrict__ Cptr, int ldc,
               int K)
{
    const int lane = threadIdx.x & 31;
    const int wave = threadIdx.x >> 5;
    const int wm   = wave & 3;       // 0..3  (M direction)
    const int wn   = wave >> 2;      // 0..1  (N direction)
    const int lr   = lane & 15;
    const int lh   = lane >> 4;      // 0/1

    const int M0 = blockIdx.y * 128 + wm * 32;
    const int N0 = blockIdx.x * 128 + wn * 64;

    // A fragment bases: lane = M row, 16 contiguous K at lh*16
    const __bf16* a0 = A + (size_t)(M0 + lr) * lda + lh * 16;
    const __bf16* a1 = a0 + (size_t)16 * lda;
    // B fragment bases: lane = N column of W, 16 contiguous K at lh*16
    const __bf16* b0 = W + (size_t)(N0 + lr) * ldw + lh * 16;
    const __bf16* b1 = b0 + (size_t)16 * ldw;
    const __bf16* b2 = b0 + (size_t)32 * ldw;
    const __bf16* b3 = b0 + (size_t)48 * ldw;

    f32x8 acc[2][4] = {};

    for (int k = 0; k < K; k += 32) {
        bf16x8 alo0 = *(const bf16x8*)(a0 + k);
        bf16x8 ahi0 = *(const bf16x8*)(a0 + k + 8);
        bf16x8 alo1 = *(const bf16x8*)(a1 + k);
        bf16x8 ahi1 = *(const bf16x8*)(a1 + k + 8);
        bf16x16 af0 = __builtin_shufflevector(alo0, ahi0,
            0,1,2,3,4,5,6,7,8,9,10,11,12,13,14,15);
        bf16x16 af1 = __builtin_shufflevector(alo1, ahi1,
            0,1,2,3,4,5,6,7,8,9,10,11,12,13,14,15);

        bf16x16 bf0 = *(const bf16x16*)(b0 + k);
        bf16x16 bf1 = *(const bf16x16*)(b1 + k);
        bf16x16 bf2 = *(const bf16x16*)(b2 + k);
        bf16x16 bf3 = *(const bf16x16*)(b3 + k);

        acc[0][0] = __builtin_amdgcn_wmma_f32_16x16x32_bf16(false, af0, false, bf0, (short)0, acc[0][0], false, false);
        acc[0][1] = __builtin_amdgcn_wmma_f32_16x16x32_bf16(false, af0, false, bf1, (short)0, acc[0][1], false, false);
        acc[0][2] = __builtin_amdgcn_wmma_f32_16x16x32_bf16(false, af0, false, bf2, (short)0, acc[0][2], false, false);
        acc[0][3] = __builtin_amdgcn_wmma_f32_16x16x32_bf16(false, af0, false, bf3, (short)0, acc[0][3], false, false);
        acc[1][0] = __builtin_amdgcn_wmma_f32_16x16x32_bf16(false, af1, false, bf0, (short)0, acc[1][0], false, false);
        acc[1][1] = __builtin_amdgcn_wmma_f32_16x16x32_bf16(false, af1, false, bf1, (short)0, acc[1][1], false, false);
        acc[1][2] = __builtin_amdgcn_wmma_f32_16x16x32_bf16(false, af1, false, bf2, (short)0, acc[1][2], false, false);
        acc[1][3] = __builtin_amdgcn_wmma_f32_16x16x32_bf16(false, af1, false, bf3, (short)0, acc[1][3], false, false);
    }

    // Epilogue. D layout: lanes 0-15: n=lane, VGPR r -> M=r;
    //                     lanes 16-31: n=lane-16, VGPR r -> M=r+8.
#pragma unroll
    for (int i = 0; i < 2; ++i) {
        const int mbase = M0 + i * 16 + lh * 8;
#pragma unroll
        for (int j = 0; j < 4; ++j) {
            const int n  = N0 + j * 16 + lr;
            const float bv = bias[n];
#pragma unroll
            for (int r = 0; r < 8; ++r) {
                float v = acc[i][j][r] + bv;
                if (MODE == 1) v = eluf(v);
                if (MODE == 0)
                    ((float*)Cptr)[(size_t)(mbase + r) * ldc + n] = v;
                else
                    ((__bf16*)Cptr)[(size_t)(mbase + r) * ldc + n] = (__bf16)v;
            }
        }
    }
}

// ============================================================
// Helpers
// ============================================================
__global__ void cvt_f32_bf16(const float* __restrict__ in,
                             __bf16* __restrict__ out, int n)
{
    int i = blockIdx.x * blockDim.x + threadIdx.x;
    if (i < n) out[i] = (__bf16)in[i];
}

// x [B,512] f32 -> z cols 128..639 (bf16, ld 640)
__global__ void pack_x_to_z(const float* __restrict__ x,
                            __bf16* __restrict__ z)
{
    int i = blockIdx.x * blockDim.x + threadIdx.x;   // one thread per 4 elems
    int total = kB * (kIN / 4);
    if (i >= total) return;
    int r  = i / (kIN / 4);
    int c4 = (i % (kIN / 4)) * 4;
    float4 v = *(const float4*)(x + (size_t)r * kIN + c4);
    bf16x4 o = { (__bf16)v.x, (__bf16)v.y, (__bf16)v.z, (__bf16)v.w };
    *(bf16x4*)(z + (size_t)r * kZK + kLAT + c4) = o;
}

__global__ void zero_f32(float* __restrict__ p, int n)
{
    int i = blockIdx.x * blockDim.x + threadIdx.x;
    if (i < n) p[i] = 0.f;
}

// Column sums over H [B,N] f32. grid = (N/64, 64), block = 256.
__global__ void col_stats(const float* __restrict__ H, int N,
                          float* __restrict__ sum, float* __restrict__ sumsq)
{
    __shared__ float ls[2][4][64];
    const int tx  = threadIdx.x & 63;
    const int ty  = threadIdx.x >> 6;           // 0..3
    const int col = blockIdx.x * 64 + tx;
    const int row0 = blockIdx.y * (kB / 64) + ty;
    float s = 0.f, q = 0.f;
    for (int r = 0; r < kB / 64; r += 4) {      // 512 rows per y-split
        float v = H[(size_t)(row0 + r) * N + col];
        s += v; q += v * v;
    }
    ls[0][ty][tx] = s; ls[1][ty][tx] = q;
    __syncthreads();
    if (ty == 0) {
        s = ls[0][0][tx] + ls[0][1][tx] + ls[0][2][tx] + ls[0][3][tx];
        q = ls[1][0][tx] + ls[1][1][tx] + ls[1][2][tx] + ls[1][3][tx];
        atomicAdd(&sum[col], s);
        atomicAdd(&sumsq[col], q);
    }
}

__global__ void bn_finalize(const float* __restrict__ sum,
                            const float* __restrict__ sumsq,
                            const float* __restrict__ g,
                            const float* __restrict__ be,
                            float* __restrict__ scale,
                            float* __restrict__ shift, int N)
{
    int c = blockIdx.x * blockDim.x + threadIdx.x;
    if (c >= N) return;
    const float invB = 1.0f / (float)kB;
    float m   = sum[c] * invB;
    float var = fmaxf(sumsq[c] * invB - m * m, 0.f);   // biased var (jnp.var)
    float sc  = g[c] * rsqrtf(var + kEPS);
    scale[c] = sc;
    shift[c] = be[c] - m * sc;
}

// out = bf16(ELU(H*scale[c] + shift[c])), 4-wide
__global__ void bn_elu_apply(const float* __restrict__ H,
                             const float* __restrict__ scale,
                             const float* __restrict__ shift,
                             __bf16* __restrict__ out,
                             int N, long total4)
{
    long i = (long)blockIdx.x * blockDim.x + threadIdx.x;
    if (i >= total4) return;
    long base = i * 4;
    int  c = (int)(base % N);          // N % 4 == 0 -> stays in-row
    float4 h = *(const float4*)(H + base);
    float v0 = eluf(h.x * scale[c + 0] + shift[c + 0]);
    float v1 = eluf(h.y * scale[c + 1] + shift[c + 1]);
    float v2 = eluf(h.z * scale[c + 2] + shift[c + 2]);
    float v3 = eluf(h.w * scale[c + 3] + shift[c + 3]);
    bf16x4 o = { (__bf16)v0, (__bf16)v1, (__bf16)v2, (__bf16)v3 };
    *(bf16x4*)(out + base) = o;
}

// Final: per row pick expert g, dot(T2[g,row,:], w3[g,:]) + b3[g],
// pred_log = -softplus, pred_prob = 10^pred_log.
__global__ void head_final(const __bf16* __restrict__ T2,   // [3][B][HALF]
                           const int* __restrict__ gidx,
                           const float* __restrict__ w3,    // [3][HALF]
                           const float* __restrict__ b3,    // [3]
                           float* __restrict__ out)
{
    const int lane = threadIdx.x & 31;
    const int row  = blockIdx.x * (blockDim.x >> 5) + (threadIdx.x >> 5);
    if (row >= kB) return;
    const int g = gidx[row];
    const __bf16* t = T2 + ((size_t)g * kB + row) * kHALF + lane * 8;
    const float*  w = w3 + g * kHALF + lane * 8;
    bf16x8 tv = *(const bf16x8*)t;
    float s = 0.f;
#pragma unroll
    for (int e = 0; e < 8; ++e) s += (float)tv[e] * w[e];
#pragma unroll
    for (int o = 16; o; o >>= 1) s += __shfl_xor(s, o, 32);
    if (lane == 0) {
        float p  = s + b3[g];
        float sp = p > 0.f ? p + log1pf(expf(-p)) : log1pf(expf(p));
        float pl = -sp;
        out[row]      = pl;
        out[kB + row] = expf(pl * kLN10);
    }
}

// ============================================================
// Host orchestration
// ============================================================
extern "C" void kernel_launch(void* const* d_in, const int* in_sizes, int n_in,
                              void* d_out, int out_size, void* d_ws, size_t ws_size,
                              hipStream_t stream)
{
    (void)in_sizes; (void)n_in; (void)out_size; (void)ws_size;

    const float* x     = (const float*)d_in[0];
    const int*   gidx  = (const int*)  d_in[1];
    const float* e_w1  = (const float*)d_in[2];
    const float* e_b1  = (const float*)d_in[3];
    const float* e_g1  = (const float*)d_in[4];
    const float* e_be1 = (const float*)d_in[5];
    const float* e_w2  = (const float*)d_in[6];
    const float* e_b2  = (const float*)d_in[7];
    const float* e_g2  = (const float*)d_in[8];
    const float* e_be2 = (const float*)d_in[9];
    const float* e_w3  = (const float*)d_in[10];
    const float* e_b3  = (const float*)d_in[11];
    const float* s_w1  = (const float*)d_in[12];
    const float* s_b1  = (const float*)d_in[13];
    const float* s_g1  = (const float*)d_in[14];
    const float* s_be1 = (const float*)d_in[15];
    const float* s_w2  = (const float*)d_in[16];
    const float* s_b2  = (const float*)d_in[17];
    const float* s_g2  = (const float*)d_in[18];
    const float* s_be2 = (const float*)d_in[19];
    const float* h_w1  = (const float*)d_in[20];
    const float* h_b1  = (const float*)d_in[21];
    const float* h_w2  = (const float*)d_in[22];
    const float* h_b2  = (const float*)d_in[23];
    const float* h_w3  = (const float*)d_in[24];
    const float* h_b3  = (const float*)d_in[25];
    float* out = (float*)d_out;

    // ---- bf16 weight pool layout (element offsets) ----
    constexpr size_t o_we1 = 0;
    constexpr size_t o_we2 = o_we1 + (size_t)kHID * kIN;
    constexpr size_t o_we3 = o_we2 + (size_t)kMID * kHID;
    constexpr size_t o_ws1 = o_we3 + (size_t)kLAT * kMID;
    constexpr size_t o_ws2 = o_ws1 + (size_t)kHID * kZK;
    constexpr size_t o_wh1 = o_ws2 + (size_t)kMID * kHID;
    constexpr size_t o_wh2 = o_wh1 + (size_t)3 * kEH * kMID;
    constexpr size_t kWB   = o_wh2 + (size_t)3 * kHALF * kEH;

    // ---- workspace carve ----
    char* wsp = (char*)d_ws;
    size_t off = 0;
    auto carve = [&](size_t bytes) -> char* {
        char* p = wsp + off;
        off += (bytes + 255) & ~(size_t)255;
        return p;
    };
    __bf16* wb     = (__bf16*)carve(kWB * 2);                     // ~17 MB
    __bf16* zbuf   = (__bf16*)carve((size_t)kB * kZK * 2);        // 40 MB
    float*  Hbuf   = (float*) carve((size_t)kB * kHID * 4);       // 256 MB
    __bf16* Abuf1  = (__bf16*)carve((size_t)kB * kHID * 2);       // 128 MB (A1/A4/T1[3])
    __bf16* Abuf2  = (__bf16*)carve((size_t)kB * kMID * 2);       // 64 MB  (A2/hidden/T2[3])
    float*  colsum = (float*) carve(2048 * 4);
    float*  colsq  = (float*) carve(2048 * 4);
    float*  scaleb = (float*) carve(2048 * 4);
    float*  shiftb = (float*) carve(2048 * 4);

    // ---- 1. weights -> bf16 ----
    struct { const float* s; size_t o; int n; } cv[7] = {
        { e_w1, o_we1, kHID * kIN },
        { e_w2, o_we2, kMID * kHID },
        { e_w3, o_we3, kLAT * kMID },
        { s_w1, o_ws1, kHID * kZK },
        { s_w2, o_ws2, kMID * kHID },
        { h_w1, o_wh1, 3 * kEH * kMID },
        { h_w2, o_wh2, 3 * kHALF * kEH },
    };
    for (int i = 0; i < 7; ++i)
        cvt_f32_bf16<<<dim3((cv[i].n + 255) / 256), 256, 0, stream>>>(
            cv[i].s, wb + cv[i].o, cv[i].n);

    // ---- 2. x -> z[:,128:640] bf16 ----
    {
        int tot = kB * (kIN / 4);
        pack_x_to_z<<<dim3((tot + 255) / 256), 256, 0, stream>>>(x, zbuf);
    }

    dim3 blk(256);
    auto run_bn = [&](int N, const float* g, const float* be, __bf16* outA) {
        zero_f32<<<dim3((4096 + 255) / 256), blk, 0, stream>>>(colsum, 2048);
        zero_f32<<<dim3((4096 + 255) / 256), blk, 0, stream>>>(colsq, 2048);
        col_stats<<<dim3(N / 64, 64), blk, 0, stream>>>(Hbuf, N, colsum, colsq);
        bn_finalize<<<dim3((N + 255) / 256), blk, 0, stream>>>(
            colsum, colsq, g, be, scaleb, shiftb, N);
        long t4 = (long)kB * N / 4;
        bn_elu_apply<<<dim3((unsigned)((t4 + 255) / 256)), blk, 0, stream>>>(
            Hbuf, scaleb, shiftb, outA, N, t4);
    };

    // ---- encoder ----
    // G1: [B,512] x [2048,512]^T -> H f32
    gemm_bf16_wmma<0><<<dim3(kHID / 128, kB / 128), blk, 0, stream>>>(
        zbuf + kLAT, kZK, wb + o_we1, kIN, e_b1, Hbuf, kHID, kIN);
    run_bn(kHID, e_g1, e_be1, Abuf1);

    // G2: [B,2048] x [1024,2048]^T
    gemm_bf16_wmma<0><<<dim3(kMID / 128, kB / 128), blk, 0, stream>>>(
        Abuf1, kHID, wb + o_we2, kHID, e_b2, Hbuf, kMID, kHID);
    run_bn(kMID, e_g2, e_be2, Abuf2);

    // G3: latent -> z[:,0:128] bf16 (bias only)
    gemm_bf16_wmma<2><<<dim3(kLAT / 128, kB / 128), blk, 0, stream>>>(
        Abuf2, kMID, wb + o_we3, kMID, e_b3, zbuf, kZK, kMID);

    // ---- shared trunk ----
    gemm_bf16_wmma<0><<<dim3(kHID / 128, kB / 128), blk, 0, stream>>>(
        zbuf, kZK, wb + o_ws1, kZK, s_b1, Hbuf, kHID, kZK);
    run_bn(kHID, s_g1, s_be1, Abuf1);

    gemm_bf16_wmma<0><<<dim3(kMID / 128, kB / 128), blk, 0, stream>>>(
        Abuf1, kHID, wb + o_ws2, kHID, s_b2, Hbuf, kMID, kHID);
    run_bn(kMID, s_g2, s_be2, Abuf2);          // -> hidden (bf16)

    // ---- expert heads (all 3, like the reference) ----
    for (int e = 0; e < 3; ++e) {              // T1[e] = ELU(hidden @ h_w1[e]^T + b1)
        gemm_bf16_wmma<1><<<dim3(kEH / 128, kB / 128), blk, 0, stream>>>(
            Abuf2, kMID, wb + o_wh1 + (size_t)e * kEH * kMID, kMID,
            h_b1 + e * kEH, Abuf1 + (size_t)e * kB * kEH, kEH, kMID);
    }
    for (int e = 0; e < 3; ++e) {              // T2[e] = ELU(T1[e] @ h_w2[e]^T + b2)
        gemm_bf16_wmma<1><<<dim3(kHALF / 128, kB / 128), blk, 0, stream>>>(
            Abuf1 + (size_t)e * kB * kEH, kEH,
            wb + o_wh2 + (size_t)e * kHALF * kEH, kEH,
            h_b2 + e * kHALF, Abuf2 + (size_t)e * kB * kHALF, kHALF, kEH);
    }

    // ---- final select + softplus ----
    head_final<<<dim3(kB / 8), blk, 0, stream>>>(Abuf2, gidx, h_w3, h_b3, out);
}
